// KMeans_69509750718469
// MI455X (gfx1250) — matrix operand order repaired and checked by reading, not compiled, not run on hardware
//
#include <hip/hip_runtime.h>
#include <hip/hip_bf16.h>
#include <stdint.h>

#define EPS 1e-6f

typedef __attribute__((ext_vector_type(16))) __bf16 v16bf;
typedef __attribute__((ext_vector_type(8)))  float  v8f;

#define DIM       512
#define NCENT     2048
#define NROWS     32768
#define NTILES    (NCENT / 16)   // 128 column tiles
#define KCHUNKS   (DIM / 32)     // 16 wmma K-steps
#define ROWS_PER_WG 128
#define THREADS   256
#define TILE_DW   4096                     // dwords per packed B tile (16 KB)
#define PACKED_DWORDS (NTILES * TILE_DW)   // 2 MB of pre-packed bf16 B tiles

// ---------------------------------------------------------------------------
// Kernel 1: convert centroids f32 -> bf16, laid out directly in this file's
// contiguous-k B-operand order:  tile t, k-chunk kc, lane l -> 16 consecutive bf16
// (lane 0-15: K = kc*32+0..15 of column n=l ; lane 16-31: K = kc*32+16..31).
// GEMM kernel then stages each tile into LDS with a raw contiguous copy.
// ---------------------------------------------------------------------------
__global__ void pack_centroids_kernel(const float* __restrict__ cent,
                                      uint32_t* __restrict__ packed) {
  int g  = blockIdx.x * blockDim.x + threadIdx.x;   // dword index 0..524287
  int t  = g >> 12;          // 4096 dwords per tile (16KB)
  int d  = g & 4095;
  int kc = d >> 8;           // 256 dwords per k-chunk
  int l  = (d >> 3) & 31;    // lane within chunk (8 dwords each)
  int jp = d & 7;            // dword within lane's 32B
  int k  = kc * 32 + ((l & 16) ? 16 : 0) + jp * 2;
  int n  = t * 16 + (l & 15);
  float f0 = cent[(size_t)n * DIM + k];
  float f1 = cent[(size_t)n * DIM + k + 1];
  union { __bf16 h[2]; uint32_t u; } p;
  p.h[0] = (__bf16)f0;
  p.h[1] = (__bf16)f1;
  packed[g] = p.u;
}

// ---------------------------------------------------------------------------
// Kernel 2: c_sq[n] = sum_k centroids[n,k]^2   (wave32 shuffle reduction)
// ---------------------------------------------------------------------------
__global__ void csq_kernel(const float* __restrict__ cent,
                           float* __restrict__ csq) {
  int row  = blockIdx.x * 8 + (threadIdx.x >> 5);
  int lane = threadIdx.x & 31;
  const float* cr = cent + (size_t)row * DIM;
  float s = 0.f;
  for (int k = lane; k < DIM; k += 32) { float v = cr[k]; s += v * v; }
  #pragma unroll
  for (int m = 16; m >= 1; m >>= 1) s += __shfl_xor(s, m, 32);
  if (lane == 0) csq[row] = s;
}

// ---------------------------------------------------------------------------
// Kernel 3: distance GEMM on bf16 WMMA + running argmin + centroid gather.
// 8 waves per WG; each wave owns 16 rows with the full A tile (16x512 bf16,
// +EPS) resident in 128 VGPRs in contiguous-k A layout. WG sweeps 128 column tiles.
// B tiles are software-pipelined through a double-buffered LDS stage.
// Inside a tile: two independent accumulator chains (even/odd k-chunks) AND
// two rotating B-operand register buffers, so 4 ds_loads are in flight ahead
// of each WMMA pair and LDS latency hides under two WMMAs.
// ---------------------------------------------------------------------------
__global__ __launch_bounds__(THREADS)
void kmeans_assign_kernel(const float* __restrict__ x,
                          const float* __restrict__ cent,
                          const uint32_t* __restrict__ packedB,
                          const float* __restrict__ c_sq,
                          float* __restrict__ labels,
                          float* __restrict__ assigned) {
  __shared__ __align__(16) uint32_t ldsB[2 * TILE_DW];   // 2 x 16 KB B tiles
  __shared__ int bestIdxLds[ROWS_PER_WG];

  const int tid    = threadIdx.x;
  const int lane   = tid & 31;
  const int waveId = tid >> 5;
  const int wgRow  = blockIdx.x * ROWS_PER_WG;
  const int hi     = (lane & 16) ? 1 : 0;
  const int r      = wgRow + waveId * 16 + (lane & 15);

  // ---- load + convert A once, resident in VGPRs (xe = x + eps) ----
  // A layout per 32-K chunk: lanes 0-15 hold K {k0..k0+15},
  // lanes 16-31 hold K {k0+16..k0+31} for the same 16 rows.
  v16bf A[KCHUNKS];
  const float* xrow = x + (size_t)r * DIM;
  #pragma unroll
  for (int kc = 0; kc < KCHUNKS; ++kc) {
    const int k0 = kc * 32 + hi * 16;
    v16bf a;
    #pragma unroll
    for (int j = 0; j < 8; ++j) a[j]     = (__bf16)(xrow[k0 + j]      + EPS);
    #pragma unroll
    for (int j = 0; j < 8; ++j) a[8 + j] = (__bf16)(xrow[k0 + 8 + j] + EPS);
    A[kc] = a;
  }

  float bestv[8];
  int   besti[8];
  #pragma unroll
  for (int i = 0; i < 8; ++i) { bestv[i] = 3.4e38f; besti[i] = 0; }

  // ---- prologue: stage tile 0 into buffer 0 ----
  {
    const float4* src = (const float4*)packedB;
    float4* dst = (float4*)ldsB;
    #pragma unroll
    for (int i = 0; i < 4; ++i) dst[tid + i * THREADS] = src[tid + i * THREADS];
  }
  __syncthreads();

  for (int t = 0; t < NTILES; ++t) {
    // issue next tile's global loads early (branch-free; last iter reloads t=0)
    const int tn = (t + 1) & (NTILES - 1);
    const float4* src = (const float4*)(packedB + (size_t)tn * TILE_DW);
    float4 pf0 = src[tid];
    float4 pf1 = src[tid + 1 * THREADS];
    float4 pf2 = src[tid + 2 * THREADS];
    float4 pf3 = src[tid + 3 * THREADS];

    // WMMA on the current buffer: dual accumulators + rotating B operands
    const char* buf = (const char*)(ldsB + (t & 1) * TILE_DW) + lane * 32;
    v8f acc0 = {};
    v8f acc1 = {};
    v16bf bA = *(const v16bf*)(buf + 0 * 1024);
    v16bf bB = *(const v16bf*)(buf + 1 * 1024);
    #pragma unroll
    for (int kc = 0; kc < KCHUNKS; kc += 2) {
      // issue loads for the next chunk pair before consuming this one
      // (wraps on the last iteration; harmless in-bounds reload of kc=0,1)
      const v16bf bC = *(const v16bf*)(buf + (((kc + 2) & (KCHUNKS - 1)) * 1024));
      const v16bf bD = *(const v16bf*)(buf + (((kc + 3) & (KCHUNKS - 1)) * 1024));
      acc0 = __builtin_amdgcn_wmma_f32_16x16x32_bf16(false, A[kc],     false, bA,
                                                     (short)0, acc0, false, false);
      acc1 = __builtin_amdgcn_wmma_f32_16x16x32_bf16(false, A[kc + 1], false, bB,
                                                     (short)0, acc1, false, false);
      bA = bC;
      bB = bD;
    }

    // dist^2 up to a row-constant:  c_sq[n] - 2 * (xe . c)
    const float csq  = c_sq[t * 16 + (lane & 15)];
    const int   nIdx = t * 16 + (lane & 15);
    #pragma unroll
    for (int i = 0; i < 8; ++i) {
      float s = fmaf(-2.0f, acc0[i] + acc1[i], csq);
      if (s < bestv[i]) { bestv[i] = s; besti[i] = nIdx; }
    }

    // drain prefetch registers into the alternate buffer, then one barrier
    float4* dst = (float4*)(ldsB + (tn & 1) * TILE_DW);
    dst[tid]               = pf0;
    dst[tid + 1 * THREADS] = pf1;
    dst[tid + 2 * THREADS] = pf2;
    dst[tid + 3 * THREADS] = pf3;
    __syncthreads();
  }

  // ---- argmin across the 16 lanes sharing each row (C layout:
  //      VGPR i, lanes 0-15 -> row i ; lanes 16-31 -> row i+8) ----
  #pragma unroll
  for (int i = 0; i < 8; ++i) {
    float bv = bestv[i];
    int   bi = besti[i];
    #pragma unroll
    for (int mask = 8; mask >= 1; mask >>= 1) {
      float ov = __shfl_xor(bv, mask, 32);
      int   oi = __shfl_xor(bi, mask, 32);
      if (ov < bv || (ov == bv && oi < bi)) { bv = ov; bi = oi; }
    }
    if ((lane & 15) == 0) {
      int rowLocal = waveId * 16 + i + hi * 8;
      labels[wgRow + rowLocal] = (float)bi;
      bestIdxLds[rowLocal] = bi;
    }
  }
  __syncthreads();

  // ---- gather assigned centroid rows (coalesced float4 copy) ----
  const float4* cent4 = (const float4*)cent;
  float4* out4 = (float4*)assigned;
  #pragma unroll
  for (int it = 0; it < (ROWS_PER_WG * (DIM / 4)) / THREADS; ++it) {
    int idx = tid + it * THREADS;
    int row = idx >> 7;          // 128 float4 per row
    int col = idx & 127;
    int c   = bestIdxLds[row];
    out4[(size_t)(wgRow + row) * (DIM / 4) + col] =
        cent4[(size_t)c * (DIM / 4) + col];
  }
}

// ---------------------------------------------------------------------------
extern "C" void kernel_launch(void* const* d_in, const int* in_sizes, int n_in,
                              void* d_out, int out_size, void* d_ws, size_t ws_size,
                              hipStream_t stream) {
  const float* x    = (const float*)d_in[0];   // (8,4096,512) f32
  const float* cent = (const float*)d_in[1];   // (2048,512)  f32

  uint32_t* packed = (uint32_t*)d_ws;                                   // 2 MB
  float*    csq    = (float*)((char*)d_ws + (size_t)PACKED_DWORDS * 4); // 8 KB

  float* labels   = (float*)d_out;           // 32768 labels (as float)
  float* assigned = (float*)d_out + NROWS;   // 32768 x 512 assigned centroids

  pack_centroids_kernel<<<PACKED_DWORDS / 256, 256, 0, stream>>>(cent, packed);
  csq_kernel<<<NCENT / 8, 256, 0, stream>>>(cent, csq);
  kmeans_assign_kernel<<<NROWS / ROWS_PER_WG, THREADS, 0, stream>>>(
      x, cent, packed, csq, labels, assigned);
}
